// MOE_80015240724877
// MI455X (gfx1250) — matrix operand order repaired and checked by reading, not compiled, not run on hardware
//
#include <hip/hip_runtime.h>
#include <math.h>

// ---------------- types ----------------
typedef __bf16 bf16_t;
typedef __attribute__((ext_vector_type(16))) __bf16 v16bf;
typedef __attribute__((ext_vector_type(8)))  float   v8f;
typedef __attribute__((ext_vector_type(8)))  __bf16  bf16x8;

#define CAT16(lo, hi) __builtin_shufflevector(lo, hi, 0,1,2,3,4,5,6,7,8,9,10,11,12,13,14,15)

// ---------------- constants ----------------
#define Lc  2
#define Hh  12
#define Cc  768
#define HSs 64
#define Ee  8
#define Vv  32000
#define Bb  2
#define Tt  1024
#define Nt  (Bb*Tt)     // 2048 tokens
#define Ff  (4*Cc)      // 3072

// =====================================================================
// Tiled bf16 WMMA GEMM:  C = alpha * A (MxK) * B + bias + resid, opt relu
//   A row-major [M,K].  TRANS_B=0: B is [K,N] row-major. TRANS_B=1: B is [N,K].
//   grid = (N/64, M/128, z). 256 threads = 8 wave32 waves, 4(M)x2(N) wave grid,
//   each wave owns a 32x32 output tile -> 4x v_wmma_f32_16x16x32_bf16 per K-step
//   with 2x fragment reuse. LDS tiles: A[128][32] row-major, B[64][32] N-major,
//   rows padded to 40 bf16 (80B, 16B-aligned, conflict-free 16-row strides).
//   Fragment gathers are pure ds_load_b128 (2 per fragment, ISA 7.12.2 layout).
//   Requires M % 128 == 0, N % 64 == 0, K % 32 == 0 (true for every call here).
// =====================================================================
template<bool TRANS_B>
__global__ void __launch_bounds__(256)
gemm_wmma(const bf16_t* __restrict__ A, int lda, long zsA,
          const bf16_t* __restrict__ Bm, int ldb, long zsB,
          float* Cf, bf16_t* Cbf, int ldc, long zsC,
          const float* __restrict__ bias, long zsBias,
          const float* resid, int ldr,
          int N, int K, float alpha, int relu)
{
    __shared__ bf16_t As[128][40];   // 10.0 KB, row-major [m][k]
    __shared__ bf16_t Bs[64][40];    //  5.0 KB, N-major  [n][k]

    const int tid  = threadIdx.x;
    const int lane = tid & 31;
    const int wave = tid >> 5;
    const int wm   = wave >> 1;      // 0..3  (32-row slice)
    const int wn   = wave & 1;       // 0..1  (32-col slice)
    const int z    = blockIdx.z;

    A  += (long)z * zsA;
    Bm += (long)z * zsB;
    if (Cf)   Cf   += (long)z * zsC;
    if (Cbf)  Cbf  += (long)z * zsC;
    if (bias) bias += (long)z * zsBias;

    const int m0 = blockIdx.y * 128;
    const int n0 = blockIdx.x * 64;

    v8f acc00 = {}, acc01 = {}, acc10 = {}, acc11 = {};

    // staging assignments: A 128x32 -> 16 bf16/thread; B 64x32 -> 8 bf16/thread
    const int a_row = tid >> 1;            // 0..127
    const int a_col = (tid & 1) * 16;      // 0 or 16
    const int bt_n  = tid & 63;            // TRANS_B==1: row of B = n
    const int bt_k  = (tid >> 6) * 8;      //             k chunk
    const int bn_k  = tid >> 3;            // TRANS_B==0: row of B = k
    const int bn_n  = (tid & 7) * 8;       //             n chunk

    const bf16_t* aptr = A + (long)(m0 + a_row) * lda;

    // ---- software pipeline: preload tile 0 into registers ----
    bf16x8 av0 = *(const bf16x8*)(aptr + a_col);
    bf16x8 av1 = *(const bf16x8*)(aptr + a_col + 8);
    bf16x8 bv;
    if (TRANS_B) bv = *(const bf16x8*)(Bm + (long)(n0 + bt_n) * ldb + bt_k);
    else         bv = *(const bf16x8*)(Bm + (long)bn_k * ldb + n0 + bn_n);

    for (int k0 = 0; k0 < K; k0 += 32) {
        __syncthreads();   // previous iteration's compute finished
        *(bf16x8*)(&As[a_row][a_col])     = av0;
        *(bf16x8*)(&As[a_row][a_col + 8]) = av1;
        if (TRANS_B) {
            *(bf16x8*)(&Bs[bt_n][bt_k]) = bv;
        } else {
            #pragma unroll
            for (int j = 0; j < 8; ++j) Bs[bn_n + j][bn_k] = bv[j];   // transpose scatter
        }
        const int k1 = k0 + 32;
        if (k1 < K) {
            av0 = *(const bf16x8*)(aptr + k1 + a_col);
            av1 = *(const bf16x8*)(aptr + k1 + a_col + 8);
            if (TRANS_B) bv = *(const bf16x8*)(Bm + (long)(n0 + bt_n) * ldb + k1 + bt_k);
            else         bv = *(const bf16x8*)(Bm + (long)(k1 + bn_k) * ldb + n0 + bn_n);
            if (k1 + 32 < K) __builtin_prefetch(aptr + k1 + 32 + a_col, 0, 1);
        }
        __syncthreads();

        // ---- fragment gathers: 8x ds_load_b128 (ISA 7.12.2 layouts) ----
        const int kg  = (lane >> 4) << 3;            // A: 0 or 8
        const int kb  = (lane >> 4) << 3;            // B: 0 or 8
        const int ar0 = (wm << 5) + (lane & 15);
        const int ar1 = ar0 + 16;
        const int c0  = (wn << 5) + (lane & 15);
        const int c1  = c0 + 16;
        bf16x8 a0lo = *(const bf16x8*)(&As[ar0][kg]);
        bf16x8 a0hi = *(const bf16x8*)(&As[ar0][16 + kg]);
        bf16x8 a1lo = *(const bf16x8*)(&As[ar1][kg]);
        bf16x8 a1hi = *(const bf16x8*)(&As[ar1][16 + kg]);
        bf16x8 b0lo = *(const bf16x8*)(&Bs[c0][kb]);
        bf16x8 b0hi = *(const bf16x8*)(&Bs[c0][16 + kb]);
        bf16x8 b1lo = *(const bf16x8*)(&Bs[c1][kb]);
        bf16x8 b1hi = *(const bf16x8*)(&Bs[c1][16 + kb]);
        v16bf af0 = CAT16(a0lo, a0hi);
        v16bf af1 = CAT16(a1lo, a1hi);
        v16bf bf0 = CAT16(b0lo, b0hi);
        v16bf bf1 = CAT16(b1lo, b1hi);

        acc00 = __builtin_amdgcn_wmma_f32_16x16x32_bf16(false, af0, false, bf0, (short)0, acc00, false, false);
        acc01 = __builtin_amdgcn_wmma_f32_16x16x32_bf16(false, af0, false, bf1, (short)0, acc01, false, false);
        acc10 = __builtin_amdgcn_wmma_f32_16x16x32_bf16(false, af1, false, bf0, (short)0, acc10, false, false);
        acc11 = __builtin_amdgcn_wmma_f32_16x16x32_bf16(false, af1, false, bf1, (short)0, acc11, false, false);
    }

    // ---- epilogue: alpha, bias, residual, relu, f32 and/or bf16 store ----
    const int cn0 = n0 + (wn << 5) + (lane & 15);
    const int cn1 = cn0 + 16;
    const int rm0 = m0 + (wm << 5) + ((lane >> 4) << 3);
    const int rm1 = rm0 + 16;
    auto emit = [&](int cm, int cn, float r) {
        r *= alpha;
        if (bias)  r += bias[cn];
        if (resid) r += resid[(long)cm * ldr + cn];
        if (relu)  r = fmaxf(r, 0.f);
        if (Cf)  Cf [(long)cm * ldc + cn] = r;
        if (Cbf) Cbf[(long)cm * ldc + cn] = (bf16_t)r;
    };
    #pragma unroll
    for (int v = 0; v < 8; ++v) {
        emit(rm0 + v, cn0, acc00[v]);
        emit(rm0 + v, cn1, acc01[v]);
        emit(rm1 + v, cn0, acc10[v]);
        emit(rm1 + v, cn1, acc11[v]);
    }
    (void)N;
}

// =====================================================================
// small elementwise / reduction kernels
// =====================================================================
__global__ void __launch_bounds__(256)
embed_kernel(const int* __restrict__ idx, const float* __restrict__ tok,
             const float* __restrict__ pos, float* __restrict__ x, long n)
{
    long i = (long)blockIdx.x * 256 + threadIdx.x;
    if (i >= n) return;
    int  c  = (int)(i % Cc);
    long bt = i / Cc;
    int  t  = (int)(bt % Tt);
    x[i] = tok[(long)idx[bt] * Cc + c] + pos[(long)t * Cc + c];
}

__global__ void __launch_bounds__(256)
cast_bf16_kernel(const float* __restrict__ in, bf16_t* __restrict__ out, long n)
{
    long i = (long)blockIdx.x * 256 + threadIdx.x;
    if (i < n) out[i] = (bf16_t)in[i];
}

__global__ void __launch_bounds__(256)
layernorm_kernel(const float* __restrict__ x, const float* __restrict__ s,
                 const float* __restrict__ b, float* out, bf16_t* out_bf)
{
    const int row = blockIdx.x;
    const float* xr = x + (long)row * Cc;
    __shared__ float red[256];
    float sum = 0.f;
    for (int c = threadIdx.x; c < Cc; c += 256) sum += xr[c];
    red[threadIdx.x] = sum; __syncthreads();
    for (int o = 128; o > 0; o >>= 1) { if (threadIdx.x < o) red[threadIdx.x] += red[threadIdx.x + o]; __syncthreads(); }
    float mu = red[0] / Cc; __syncthreads();
    float vs = 0.f;
    for (int c = threadIdx.x; c < Cc; c += 256) { float d = xr[c] - mu; vs += d * d; }
    red[threadIdx.x] = vs; __syncthreads();
    for (int o = 128; o > 0; o >>= 1) { if (threadIdx.x < o) red[threadIdx.x] += red[threadIdx.x + o]; __syncthreads(); }
    float inv = rsqrtf(red[0] / Cc + 1e-5f);
    for (int c = threadIdx.x; c < Cc; c += 256) {
        float y = (xr[c] - mu) * inv * s[c] + b[c];
        if (out)    out   [(long)row * Cc + c] = y;
        if (out_bf) out_bf[(long)row * Cc + c] = (bf16_t)y;
    }
}

// causal softmax over one attention matrix row; writes bf16 probs (0 where masked)
__global__ void __launch_bounds__(256)
softmax_causal_kernel(const float* __restrict__ att, bf16_t* __restrict__ out)
{
    const int t = blockIdx.x;
    const float* row = att + (long)t * Tt;
    __shared__ float red[256];
    float mx = -1e30f;
    for (int s = threadIdx.x; s <= t; s += 256) mx = fmaxf(mx, row[s]);
    red[threadIdx.x] = mx; __syncthreads();
    for (int o = 128; o > 0; o >>= 1) { if (threadIdx.x < o) red[threadIdx.x] = fmaxf(red[threadIdx.x], red[threadIdx.x + o]); __syncthreads(); }
    mx = red[0]; __syncthreads();
    float sum = 0.f;
    for (int s = threadIdx.x; s <= t; s += 256) sum += expf(row[s] - mx);
    red[threadIdx.x] = sum; __syncthreads();
    for (int o = 128; o > 0; o >>= 1) { if (threadIdx.x < o) red[threadIdx.x] += red[threadIdx.x + o]; __syncthreads(); }
    float inv = 1.0f / red[0];
    for (int s = threadIdx.x; s < Tt; s += 256)
        out[(long)t * Tt + s] = (bf16_t)((s <= t) ? expf(row[s] - mx) * inv : 0.0f);
}

// top-2 of 8 gating; writes dense combine weights [Nt, Ee] (softmax over top-2)
__global__ void __launch_bounds__(256)
gate_topk_kernel(const float* __restrict__ h, const float* __restrict__ gw,
                 float* __restrict__ comb)
{
    int t = blockIdx.x * 256 + threadIdx.x;
    if (t >= Nt) return;
    const float* hr = h + (long)t * Cc;
    float g[Ee];
    for (int e = 0; e < Ee; ++e) {
        const float* w = gw + (long)e * Cc;
        float s = 0.f;
        for (int c = 0; c < Cc; ++c) s += hr[c] * w[c];
        g[e] = s;
    }
    int i1 = 0;
    for (int e = 1; e < Ee; ++e) if (g[e] > g[i1]) i1 = e;          // first max
    int i2 = -1;
    for (int e = 0; e < Ee; ++e) { if (e == i1) continue; if (i2 < 0 || g[e] > g[i2]) i2 = e; }
    float wb = expf(g[i2] - g[i1]);
    float wa = 1.0f / (1.0f + wb);
    wb *= wa;
    for (int e = 0; e < Ee; ++e)
        comb[(long)t * Ee + e] = (e == i1) ? wa : ((e == i2) ? wb : 0.0f);
}

__global__ void __launch_bounds__(256)
moe_accum_kernel(float* __restrict__ x, const float* __restrict__ y,
                 const float* __restrict__ comb, int e, long n)
{
    long i = (long)blockIdx.x * 256 + threadIdx.x;
    if (i >= n) return;
    long t = i / Cc;
    x[i] += comb[t * Ee + e] * y[i];
}

__global__ void __launch_bounds__(256)
loss_row_kernel(const float* __restrict__ logits, const int* __restrict__ targets,
                float* __restrict__ row_loss)
{
    const int r = blockIdx.x;
    const float* lr = logits + (long)r * Vv;
    __shared__ float red[256];
    float mx = -1e30f;
    for (int v = threadIdx.x; v < Vv; v += 256) mx = fmaxf(mx, lr[v]);
    red[threadIdx.x] = mx; __syncthreads();
    for (int o = 128; o > 0; o >>= 1) { if (threadIdx.x < o) red[threadIdx.x] = fmaxf(red[threadIdx.x], red[threadIdx.x + o]); __syncthreads(); }
    mx = red[0]; __syncthreads();
    float sum = 0.f;
    for (int v = threadIdx.x; v < Vv; v += 256) sum += expf(lr[v] - mx);
    red[threadIdx.x] = sum; __syncthreads();
    for (int o = 128; o > 0; o >>= 1) { if (threadIdx.x < o) red[threadIdx.x] += red[threadIdx.x + o]; __syncthreads(); }
    if (threadIdx.x == 0)
        row_loss[r] = logf(red[0]) + mx - lr[targets[r]];
}

__global__ void __launch_bounds__(256)
loss_reduce_kernel(const float* __restrict__ row_loss, float* __restrict__ out)
{
    __shared__ float red[256];
    float s = 0.f;
    for (int i = threadIdx.x; i < Nt; i += 256) s += row_loss[i];
    red[threadIdx.x] = s; __syncthreads();
    for (int o = 128; o > 0; o >>= 1) { if (threadIdx.x < o) red[threadIdx.x] += red[threadIdx.x + o]; __syncthreads(); }
    if (threadIdx.x == 0) *out = red[0] / (float)Nt;
}

// =====================================================================
// host orchestration
// =====================================================================
extern "C" void kernel_launch(void* const* d_in, const int* in_sizes, int n_in,
                              void* d_out, int out_size, void* d_ws, size_t ws_size,
                              hipStream_t stream)
{
    (void)in_sizes; (void)n_in; (void)out_size; (void)ws_size;
    const int*   idx     = (const int*)  d_in[0];
    const int*   targets = (const int*)  d_in[1];
    const float* tok_emb = (const float*)d_in[2];
    const float* pos_emb = (const float*)d_in[3];
    const float* ln1_s   = (const float*)d_in[4];
    const float* ln1_b   = (const float*)d_in[5];
    const float* wq      = (const float*)d_in[6];
    const float* wk      = (const float*)d_in[7];
    const float* wv      = (const float*)d_in[8];
    const float* wo      = (const float*)d_in[9];
    const float* bo      = (const float*)d_in[10];
    const float* ln2_s   = (const float*)d_in[11];
    const float* ln2_b   = (const float*)d_in[12];
    const float* gate_w  = (const float*)d_in[13];
    const float* w1      = (const float*)d_in[14];
    const float* b1      = (const float*)d_in[15];
    const float* w2      = (const float*)d_in[16];
    const float* b2      = (const float*)d_in[17];
    const float* lm_w    = (const float*)d_in[18];
    const float* lm_b    = (const float*)d_in[19];

    // ---- workspace arena (~120 MB) ----
    size_t cur = 0;
    char* base = (char*)d_ws;
    auto alloc = [&](size_t bytes) -> void* {
        void* p = base + cur;
        cur += (bytes + 255) & ~(size_t)255;
        return p;
    };
    float*  x      = (float*) alloc((size_t)Nt * Cc * 4);
    float*  h      = (float*) alloc((size_t)Nt * Cc * 4);
    bf16_t* h_bf   = (bf16_t*)alloc((size_t)Nt * Cc * 2);
    bf16_t* q_bf   = (bf16_t*)alloc((size_t)Bb * Hh * Tt * HSs * 2);
    bf16_t* k_bf   = (bf16_t*)alloc((size_t)Bb * Hh * Tt * HSs * 2);
    bf16_t* v_bf   = (bf16_t*)alloc((size_t)Bb * Hh * Tt * HSs * 2);
    bf16_t* o_bf   = (bf16_t*)alloc((size_t)Nt * Cc * 2);
    float*  att    = (float*) alloc((size_t)Tt * Tt * 4);
    bf16_t* att_bf = (bf16_t*)alloc((size_t)Tt * Tt * 2);
    bf16_t* wq_bf  = (bf16_t*)alloc((size_t)Hh * Cc * HSs * 2);
    bf16_t* wk_bf  = (bf16_t*)alloc((size_t)Hh * Cc * HSs * 2);
    bf16_t* wv_bf  = (bf16_t*)alloc((size_t)Hh * Cc * HSs * 2);
    bf16_t* wo_bf  = (bf16_t*)alloc((size_t)Cc * Cc * 2);
    bf16_t* w1_bf  = (bf16_t*)alloc((size_t)Ff * Cc * 2);
    bf16_t* w2_bf  = (bf16_t*)alloc((size_t)Ff * Cc * 2);
    float*  comb   = (float*) alloc((size_t)Nt * Ee * 4);
    float*  y      = (float*) alloc((size_t)Nt * Cc * 4);
    float*  rloss  = (float*) alloc((size_t)Nt * 4);
    bf16_t* x_bf   = (bf16_t*)alloc((size_t)Nt * Cc * 2);
    bf16_t* hid_bf = (bf16_t*)alloc((size_t)Nt * Ff * 2);
    bf16_t* lm_bf  = (bf16_t*)alloc((size_t)Vv * Cc * 2);

    const float attnScale = 1.0f / sqrtf((float)Cc);   // reference uses C**-0.5

    // x = tok_emb[idx] + pos_emb[:T]
    embed_kernel<<<(Nt * Cc) / 256, 256, 0, stream>>>(idx, tok_emb, pos_emb, x, (long)Nt * Cc);

    for (int l = 0; l < Lc; ++l) {
        // ---- LN1 -> h_bf ----
        layernorm_kernel<<<Nt, 256, 0, stream>>>(x, ln1_s + (size_t)l * Cc,
                                                 ln1_b + (size_t)l * Cc, nullptr, h_bf);
        // ---- per-layer weight casts ----
        long wsz = (long)Hh * Cc * HSs;
        cast_bf16_kernel<<<(wsz + 255) / 256, 256, 0, stream>>>(wq + (size_t)l * wsz, wq_bf, wsz);
        cast_bf16_kernel<<<(wsz + 255) / 256, 256, 0, stream>>>(wk + (size_t)l * wsz, wk_bf, wsz);
        cast_bf16_kernel<<<(wsz + 255) / 256, 256, 0, stream>>>(wv + (size_t)l * wsz, wv_bf, wsz);
        long wosz = (long)Cc * Cc;
        cast_bf16_kernel<<<(wosz + 255) / 256, 256, 0, stream>>>(wo + (size_t)l * wosz, wo_bf, wosz);

        // ---- QKV projections: z batches over heads, [T,C] x [C,HS] (NN) ----
        for (int b = 0; b < Bb; ++b) {
            dim3 g(HSs / 64, Tt / 128, Hh);
            gemm_wmma<false><<<g, 256, 0, stream>>>(
                h_bf + (size_t)b * Tt * Cc, Cc, 0L, wq_bf, HSs, (long)Cc * HSs,
                nullptr, q_bf + (size_t)b * Hh * Tt * HSs, HSs, (long)Tt * HSs,
                nullptr, 0L, nullptr, 0, HSs, Cc, 1.0f, 0);
            gemm_wmma<false><<<g, 256, 0, stream>>>(
                h_bf + (size_t)b * Tt * Cc, Cc, 0L, wk_bf, HSs, (long)Cc * HSs,
                nullptr, k_bf + (size_t)b * Hh * Tt * HSs, HSs, (long)Tt * HSs,
                nullptr, 0L, nullptr, 0, HSs, Cc, 1.0f, 0);
            gemm_wmma<false><<<g, 256, 0, stream>>>(
                h_bf + (size_t)b * Tt * Cc, Cc, 0L, wv_bf, HSs, (long)Cc * HSs,
                nullptr, v_bf + (size_t)b * Hh * Tt * HSs, HSs, (long)Tt * HSs,
                nullptr, 0L, nullptr, 0, HSs, Cc, 1.0f, 0);
        }

        // ---- attention per (b,h): QK^T (NT) -> causal softmax -> AV (NN) ----
        for (int b = 0; b < Bb; ++b)
            for (int hh = 0; hh < Hh; ++hh) {
                const bf16_t* qp = q_bf + ((size_t)b * Hh + hh) * Tt * HSs;
                const bf16_t* kp = k_bf + ((size_t)b * Hh + hh) * Tt * HSs;
                const bf16_t* vp = v_bf + ((size_t)b * Hh + hh) * Tt * HSs;
                gemm_wmma<true><<<dim3(Tt / 64, Tt / 128, 1), 256, 0, stream>>>(
                    qp, HSs, 0L, kp, HSs, 0L, att, nullptr, Tt, 0L,
                    nullptr, 0L, nullptr, 0, Tt, HSs, attnScale, 0);
                softmax_causal_kernel<<<Tt, 256, 0, stream>>>(att, att_bf);
                gemm_wmma<false><<<dim3(1, Tt / 128, 1), 256, 0, stream>>>(
                    att_bf, Tt, 0L, vp, HSs, 0L,
                    nullptr, o_bf + (size_t)b * Tt * Cc + (size_t)hh * HSs, Cc, 0L,
                    nullptr, 0L, nullptr, 0, HSs, Tt, 1.0f, 0);
            }

        // ---- output projection + bias + residual (in-place into x) ----
        gemm_wmma<true><<<dim3(Cc / 64, Nt / 128, 1), 256, 0, stream>>>(
            o_bf, Cc, 0L, wo_bf, Cc, 0L, x, nullptr, Cc, 0L,
            bo + (size_t)l * Cc, 0L, x, Cc, Cc, Cc, 1.0f, 0);

        // ---- LN2 -> h (f32 for gating) + h_bf ----
        layernorm_kernel<<<Nt, 256, 0, stream>>>(x, ln2_s + (size_t)l * Cc,
                                                 ln2_b + (size_t)l * Cc, h, h_bf);
        gate_topk_kernel<<<Nt / 256, 256, 0, stream>>>(h, gate_w + (size_t)l * Ee * Cc, comb);

        // ---- MoE FFN, expert loop (buffers reused; stream-serialized) ----
        for (int e = 0; e < Ee; ++e) {
            long esz = (long)Ff * Cc;
            cast_bf16_kernel<<<(esz + 255) / 256, 256, 0, stream>>>(
                w1 + ((size_t)l * Ee + e) * esz, w1_bf, esz);
            cast_bf16_kernel<<<(esz + 255) / 256, 256, 0, stream>>>(
                w2 + ((size_t)l * Ee + e) * esz, w2_bf, esz);
            // hid = relu(h @ w1^T + b1) -> bf16
            gemm_wmma<true><<<dim3(Ff / 64, Nt / 128, 1), 256, 0, stream>>>(
                h_bf, Cc, 0L, w1_bf, Cc, 0L, nullptr, hid_bf, Ff, 0L,
                b1 + ((size_t)l * Ee + e) * Ff, 0L, nullptr, 0, Ff, Cc, 1.0f, 1);
            // y = hid @ w2^T + b2
            gemm_wmma<true><<<dim3(Cc / 64, Nt / 128, 1), 256, 0, stream>>>(
                hid_bf, Ff, 0L, w2_bf, Ff, 0L, y, nullptr, Cc, 0L,
                b2 + ((size_t)l * Ee + e) * Cc, 0L, nullptr, 0, Cc, Ff, 1.0f, 0);
            // x += comb[:,e] * y
            moe_accum_kernel<<<(Nt * Cc) / 256, 256, 0, stream>>>(x, y, comb, e, (long)Nt * Cc);
        }
    }

    // ---- LM head: logits straight into d_out ----
    cast_bf16_kernel<<<((long)Nt * Cc + 255) / 256, 256, 0, stream>>>(x, x_bf, (long)Nt * Cc);
    cast_bf16_kernel<<<((long)Vv * Cc + 255) / 256, 256, 0, stream>>>(lm_w, lm_bf, (long)Vv * Cc);
    float* lf = (float*)d_out;
    gemm_wmma<true><<<dim3(Vv / 64, Nt / 128, 1), 256, 0, stream>>>(
        x_bf, Cc, 0L, lm_bf, Cc, 0L, lf, nullptr, Vv, 0L,
        lm_b, 0L, nullptr, 0, Vv, Cc, 1.0f, 0);

    // ---- cross-entropy loss ----
    loss_row_kernel<<<Nt, 256, 0, stream>>>(lf, targets, rloss);
    loss_reduce_kernel<<<1, 256, 0, stream>>>(rloss, lf + (size_t)Nt * Vv);
}
